// TransformerPriorCoder_14121852469613
// MI455X (gfx1250) — matrix operand order repaired and checked by reading, not compiled, not run on hardware
//
#include <hip/hip_runtime.h>
#include <hip/hip_bf16.h>
#include <math.h>

// ---------------------------------------------------------------------------
// TransformerPriorCoder forward for gfx1250 (MI455X).
//  - All dense GEMMs: v_wmma_f32_16x16x32_bf16, bf16 operands pre-converted
//    (weights transposed+converted once per call; activations emitted as bf16
//    by the producing kernel), f32 accumulation.
//  - GEMM staging: GLOBAL_LOAD_ASYNC_TO_LDS_B128 (ASYNCcnt) when available,
//    else b128 load + b128 LDS store. No conversion VALU in the hot loop.
//  - Attention (dk=12): VALU online softmax (negligible FLOP share).
// ---------------------------------------------------------------------------

#define D      192
#define DI     2048
#define NHEAD  16
#define DKH    12
#define BM     64
#define BN     64
#define BKK    32
#define LDT    40      // LDS row pitch (ushort) -> 80B: 16B aligned, conflict-free
#define FCH    4096    // FFN row chunk

typedef __attribute__((ext_vector_type(16))) __bf16 v16bf;
typedef __attribute__((ext_vector_type(8)))  float  v8f;
typedef int v4i_ __attribute__((vector_size(16)));   // int4 for async-LDS builtin

#define AS1 __attribute__((address_space(1)))
#define AS3 __attribute__((address_space(3)))

#if defined(__HIP_DEVICE_COMPILE__) && __has_builtin(__builtin_amdgcn_global_load_async_to_lds_b128)
#define HAVE_ASYNC_LDS 1
#else
#define HAVE_ASYNC_LDS 0
#endif

__device__ __forceinline__ unsigned short f2bf(float f) {
  return __builtin_bit_cast(unsigned short, (__bf16)f);   // hw RNE convert
}

// ---------------------------------------------------------------------------
// GEMM: C[M,N] = act(A[M,K] @ Bt[N,K]^T + bias) * scale
// A, Bt are bf16 (ushort). flags: 1=bias, 2=relu, 4=bf16 output.
// M%64==0, N%64==0, K%32==0 for every call in this model.
// ---------------------------------------------------------------------------
__global__ __launch_bounds__(256) void gemm_bf16_wmma_kernel(
    const unsigned short* __restrict__ A, const unsigned short* __restrict__ Bt,
    const float* __restrict__ bias, void* __restrict__ Cout,
    int M, int N, int K, int flags, float scale)
{
  __shared__ __align__(16) unsigned short lA[BM * LDT];
  __shared__ __align__(16) unsigned short lB[BN * LDT];

  const int tid  = threadIdx.x;
  const int bm   = blockIdx.y * BM;
  const int bn   = blockIdx.x * BN;
  const int lane = tid & 31;
  const int w    = tid >> 5;
  const int wm   = w & 3;        // M subtile (16 rows)
  const int wn   = w >> 2;       // N subtile pair (32 cols)
  const int half = lane >> 4;
  const int l16  = lane & 15;

  v8f acc0 = {};
  v8f acc1 = {};

  const int row = tid >> 2;            // 0..63
  const int kg  = (tid & 3) * 8;       // 0,8,16,24  (8 bf16 = 16B)

  const unsigned short* gA = A  + (size_t)(bm + row) * K + kg;
  const unsigned short* gB = Bt + (size_t)(bn + row) * K + kg;
  unsigned short* dA = &lA[row * LDT + kg];
  unsigned short* dB = &lB[row * LDT + kg];

  for (int k0 = 0; k0 < K; k0 += BKK) {
#if HAVE_ASYNC_LDS
    __builtin_amdgcn_global_load_async_to_lds_b128(
        (AS1 v4i_*)(AS1 void*)(void*)(gA + k0), (AS3 v4i_*)(AS3 void*)dA, 0, 0);
    __builtin_amdgcn_global_load_async_to_lds_b128(
        (AS1 v4i_*)(AS1 void*)(void*)(gB + k0), (AS3 v4i_*)(AS3 void*)dB, 0, 0);
    if (k0 + BKK < K) {
      __builtin_prefetch((const void*)(gA + k0 + BKK), 0, 1);
      __builtin_prefetch((const void*)(gB + k0 + BKK), 0, 1);
    }
#if __has_builtin(__builtin_amdgcn_s_wait_asynccnt)
    __builtin_amdgcn_s_wait_asynccnt(0);
#else
    asm volatile("s_wait_asynccnt 0x0" ::: "memory");
#endif
#else
    uint4 va = *(const uint4*)(gA + k0);
    uint4 vb = *(const uint4*)(gB + k0);
    if (k0 + BKK < K) {
      __builtin_prefetch((const void*)(gA + k0 + BKK), 0, 1);
      __builtin_prefetch((const void*)(gB + k0 + BKK), 0, 1);
    }
    *(uint4*)dA = va;
    *(uint4*)dB = vb;
#endif
    __syncthreads();

    // fragments (contiguous k = half*16 + s, same order for A and B)
    union { v16bf v; uint4 q[2]; } fa, fb0, fb1;
    const unsigned short* pA = &lA[(wm * 16 + l16) * LDT + half * 16];
    fa.q[0] = *(const uint4*)pA;                  // K = half*16 .. +7
    fa.q[1] = *(const uint4*)(pA + 8);            // K = half*16+8 .. +15
    const unsigned short* pB0 = &lB[(wn * 32 + l16) * LDT + half * 16];
    fb0.q[0] = *(const uint4*)pB0;                // K = half*16 .. +7
    fb0.q[1] = *(const uint4*)(pB0 + 8);          // K = half*16+8 .. +15
    const unsigned short* pB1 = &lB[(wn * 32 + 16 + l16) * LDT + half * 16];
    fb1.q[0] = *(const uint4*)pB1;
    fb1.q[1] = *(const uint4*)(pB1 + 8);

    acc0 = __builtin_amdgcn_wmma_f32_16x16x32_bf16(false, fa.v, false, fb0.v,
                                                   (short)0, acc0, false, false);
    acc1 = __builtin_amdgcn_wmma_f32_16x16x32_bf16(false, fa.v, false, fb1.v,
                                                   (short)0, acc1, false, false);
    __syncthreads();
  }

  const int col0 = bn + wn * 32 + l16;
  const int col1 = col0 + 16;
  const float bv0 = (flags & 1) ? bias[col0] : 0.f;
  const float bv1 = (flags & 1) ? bias[col1] : 0.f;
  const int rbase = bm + wm * 16 + half * 8;
#pragma unroll
  for (int r = 0; r < 8; ++r) {
    float v0 = acc0[r] + bv0;
    float v1 = acc1[r] + bv1;
    if (flags & 2) { v0 = fmaxf(v0, 0.f); v1 = fmaxf(v1, 0.f); }
    v0 *= scale; v1 *= scale;
    size_t o0 = (size_t)(rbase + r) * N + col0;
    size_t o1 = (size_t)(rbase + r) * N + col1;
    if (flags & 4) {
      ((unsigned short*)Cout)[o0] = f2bf(v0);
      ((unsigned short*)Cout)[o1] = f2bf(v1);
    } else {
      ((float*)Cout)[o0] = v0;
      ((float*)Cout)[o1] = v1;
    }
  }
}

// ---------------------------------------------------------------------------
// LayerNorm: X[row] = LN(Xin[row] (+Yadd) (+posenc(row%peL)))*s + b
// writes f32 residual stream AND bf16 mirror (next GEMM's A operand).
// one wave32 per row, 6 elements/lane.
// ---------------------------------------------------------------------------
__global__ void ln_kernel(float* __restrict__ Xout, unsigned short* __restrict__ Xb,
                          const float* __restrict__ Xin, const float* __restrict__ Yadd,
                          const float* __restrict__ sc, const float* __restrict__ bi,
                          int M, int peL)
{
  int row = blockIdx.x;
  if (row >= M) return;
  int lane = threadIdx.x;
  float vals[6];
  float sum = 0.f;
#pragma unroll
  for (int t = 0; t < 6; ++t) {
    int c = lane + t * 32;
    float v = Xin[(size_t)row * D + c];
    if (Yadd) v += Yadd[(size_t)row * D + c];
    if (peL > 0) {
      int pos = row % peL;
      float freq = __expf(-9.210340371976184f * (float)(c & ~1) / (float)D);
      float ang = (float)pos * freq;
      v += (c & 1) ? __cosf(ang) : __sinf(ang);
    }
    vals[t] = v;
    sum += v;
  }
#pragma unroll
  for (int o = 16; o > 0; o >>= 1) sum += __shfl_xor(sum, o, 32);
  float mean = sum * (1.f / D);
  float var = 0.f;
#pragma unroll
  for (int t = 0; t < 6; ++t) { float d = vals[t] - mean; var += d * d; }
#pragma unroll
  for (int o = 16; o > 0; o >>= 1) var += __shfl_xor(var, o, 32);
  float inv = rsqrtf(var * (1.f / D) + 1e-6f);
#pragma unroll
  for (int t = 0; t < 6; ++t) {
    int c = lane + t * 32;
    float r = (vals[t] - mean) * inv * sc[c] + bi[c];
    Xout[(size_t)row * D + c] = r;
    Xb[(size_t)row * D + c] = f2bf(r);
  }
}

// ---------------------------------------------------------------------------
// Attention: per-thread online softmax over one (batch, head, query). dk=12.
// Q/K/V f32; output written as bf16 (feeds the W_O GEMM).
// ---------------------------------------------------------------------------
__global__ void attn_kernel(const float* __restrict__ Q, const float* __restrict__ K,
                            const float* __restrict__ V, unsigned short* __restrict__ O,
                            int B, int Lq, int Lk, int causal)
{
  int idx = blockIdx.x * blockDim.x + threadIdx.x;
  int total = B * NHEAD * Lq;
  if (idx >= total) return;
  int q = idx % Lq;
  int t = idx / Lq;
  int h = t % NHEAD;
  int bb = t / NHEAD;

  const float* qp = Q + ((size_t)(bb * Lq + q)) * D + h * DKH;
  float qv[DKH];
#pragma unroll
  for (int j = 0; j < DKH; ++j) qv[j] = qp[j];

  float m = -3.0e38f, l = 0.f;
  float acc[DKH];
#pragma unroll
  for (int j = 0; j < DKH; ++j) acc[j] = 0.f;

  const float sfac = 0.2886751345948129f;  // 1/sqrt(12)
  for (int k = 0; k < Lk; ++k) {
    const float* kp = K + ((size_t)(bb * Lk + k)) * D + h * DKH;
    float s = 0.f;
#pragma unroll
    for (int j = 0; j < DKH; ++j) s += qv[j] * kp[j];
    s *= sfac;
    if (causal && k > q) s = -1e9f;
    float mn = fmaxf(m, s);
    float cf = __expf(m - mn);
    float p  = __expf(s - mn);
    l = l * cf + p;
    const float* vp = V + ((size_t)(bb * Lk + k)) * D + h * DKH;
#pragma unroll
    for (int j = 0; j < DKH; ++j) acc[j] = acc[j] * cf + p * vp[j];
    m = mn;
  }
  float inv = 1.f / l;
  unsigned short* op = O + ((size_t)(bb * Lq + q)) * D + h * DKH;
#pragma unroll
  for (int j = 0; j < DKH; ++j) op[j] = f2bf(acc[j] * inv);
}

// ---------------------------------------------------------------------------
// Conversion kernels (weights: transpose [K,N]f32 -> [N,K]bf16, once per call)
// ---------------------------------------------------------------------------
__global__ void cvt_transpose_kernel(const float* __restrict__ src,
                                     unsigned short* __restrict__ dst, int K, int N)
{
  int idx = blockIdx.x * blockDim.x + threadIdx.x;
  if (idx >= K * N) return;
  int n = idx % N, k = idx / N;               // coalesced read
  dst[(size_t)n * K + k] = f2bf(src[idx]);
}

__global__ void cvt_flat_kernel(const float* __restrict__ src,
                                unsigned short* __restrict__ dst, int n)
{
  int i = blockIdx.x * blockDim.x + threadIdx.x;
  if (i < n) dst[i] = f2bf(src[i]);
}

// ---------------------------------------------------------------------------
// Elementwise / gather-scatter kernels
// ---------------------------------------------------------------------------
__global__ void ew_outq_kernel(const float* __restrict__ f, const float* __restrict__ n,
                               float* __restrict__ o, int N)
{
  int i = blockIdx.x * blockDim.x + threadIdx.x;
  if (i < N) o[i] = f[i] + n[i] - 0.5f;
}

// overlap tokens: prev[2,192,32,32] -> Xb[8192,192] bf16 (8x8 patch, stride 4, pad 2)
__global__ void gather_overlap_kernel(const float* __restrict__ src,
                                      unsigned short* __restrict__ dst)
{
  int idx = blockIdx.x * blockDim.x + threadIdx.x;
  if (idx >= 8192 * D) return;
  int c = idx % D, row = idx / D;
  int t = row & 63, bb = row >> 6;
  int tile = bb & 63, b = bb >> 6;
  int ty = tile >> 3, tx = tile & 7;
  int py = t >> 3, px = t & 7;
  int y = ty * 4 - 2 + py, x = tx * 4 - 2 + px;
  float v = 0.f;
  if ((unsigned)y < 32u && (unsigned)x < 32u)
    v = src[(((size_t)b * D + c) * 32 + y) * 32 + x];
  dst[idx] = f2bf(v);
}

// X rows (bb*64+l) -> Zj rows (bb*128 + off + l)
__global__ void scatter_zsep_kernel(const float* __restrict__ X, float* __restrict__ Zj, int off)
{
  int idx = blockIdx.x * blockDim.x + threadIdx.x;
  if (idx >= 8192 * D) return;
  int c = idx % D, row = idx / D;
  int l = row & 63, bb = row >> 6;
  Zj[((size_t)(bb * 128 + off + l)) * D + c] = X[idx];
}

// decoder trg tokens: [start, cur_tok[0..14]] -> Xb[2048,192] bf16
__global__ void dec_tok_kernel(const float* __restrict__ outq, const float* __restrict__ start,
                               unsigned short* __restrict__ Xb)
{
  int idx = blockIdx.x * blockDim.x + threadIdx.x;
  if (idx >= 2048 * D) return;
  int c = idx % D, row = idx / D;
  int t = row & 15, bb = row >> 4;
  float v;
  if (t == 0) {
    v = start[c];
  } else {
    int tt = t - 1;
    int py = tt >> 2, px = tt & 3;
    int tile = bb & 63, b = bb >> 6;
    int ty = tile >> 3, tx = tile & 7;
    v = outq[(((size_t)b * D + c) * 32 + ty * 4 + py) * 32 + tx * 4 + px];
  }
  Xb[idx] = f2bf(v);
}

// tok[2048,384] -> condition[2,384,32,32]
__global__ void cond_scatter_kernel(const float* __restrict__ Y, float* __restrict__ cond)
{
  int idx = blockIdx.x * blockDim.x + threadIdx.x;
  if (idx >= 2 * 384 * 32 * 32) return;
  int x = idx & 31, y = (idx >> 5) & 31;
  int c = (idx >> 10) % 384, b = idx / (384 * 1024);
  int ty = y >> 2, py = y & 3, tx = x >> 2, px = x & 3;
  int row = (b * 64 + ty * 8 + tx) * 16 + py * 4 + px;
  cond[idx] = Y[(size_t)row * 384 + c];
}

// tok[2048,192] -> z_cur_f[2,192,32,32]
__global__ void zf_scatter_kernel(const float* __restrict__ X, float* __restrict__ zf)
{
  int idx = blockIdx.x * blockDim.x + threadIdx.x;
  if (idx >= 2 * D * 32 * 32) return;
  int x = idx & 31, y = (idx >> 5) & 31;
  int c = (idx >> 10) % D, b = idx / (D * 1024);
  int ty = y >> 2, py = y & 3, tx = x >> 2, px = x & 3;
  int row = (b * 64 + ty * 8 + tx) * 16 + py * 4 + px;
  zf[idx] = X[(size_t)row * D + c];
}

__global__ void lik_kernel(const float* __restrict__ outq, const float* __restrict__ cond,
                           float* __restrict__ lik)
{
  int idx = blockIdx.x * blockDim.x + threadIdx.x;
  if (idx >= 2 * D * 32 * 32) return;
  int x = idx & 31, y = (idx >> 5) & 31;
  int c = (idx >> 10) % D, b = idx / (D * 1024);
  size_t mofs = (((size_t)b * 384 + c) * 32 + y) * 32 + x;
  size_t sofs = (((size_t)b * 384 + c + 192) * 32 + y) * 32 + x;
  float mean = cond[mofs];
  float scl  = fmaxf(fabsf(cond[sofs]), 0.11f);
  float oq   = outq[idx];
  float inv  = 1.f / scl;
  float hi = (oq + 0.5f - mean) * inv;
  float lo = (oq - 0.5f - mean) * inv;
  const float kk = 0.7071067811865475f;
  float p = 0.5f * (erff(hi * kk) - erff(lo * kk));
  lik[idx] = fmaxf(p, 1e-9f);
}

// ---------------------------------------------------------------------------
// Host orchestration
// ---------------------------------------------------------------------------
static inline void gemm(hipStream_t s, const unsigned short* A, const unsigned short* Bt,
                        const float* bias, void* C, int M, int N, int K, int flags, float scale)
{
  dim3 g(N / BN, M / BM);
  gemm_bf16_wmma_kernel<<<g, 256, 0, s>>>(A, Bt, bias, C, M, N, K, flags, scale);
}
static inline void ln(hipStream_t s, float* out, unsigned short* outb, const float* in,
                      const float* add, const float* sc, const float* bi, int M, int peL)
{
  ln_kernel<<<M, 32, 0, s>>>(out, outb, in, add, sc, bi, M, peL);
}
static inline void attn(hipStream_t s, const float* Q, const float* K, const float* V,
                        unsigned short* O, int B, int Lq, int Lk, int causal)
{
  int total = B * NHEAD * Lq;
  attn_kernel<<<(total + 255) / 256, 256, 0, s>>>(Q, K, V, O, B, Lq, Lk, causal);
}

// Input flattening (recursive dict insertion order of setup_inputs()):
enum {
  IN_FEATURE = 0, IN_PREV1, IN_PREV2, IN_NOISE, IN_SZ,
  SEP_IN_S, SEP_IN_B, SEP_WQ, SEP_WK, SEP_WV, SEP_WO, SEP_LN1S, SEP_LN1B,
  SEP_W1, SEP_B1, SEP_W2, SEP_B2, SEP_LN2S, SEP_LN2B, SEP_PROJW, SEP_PROJB,
  J_IN_S, J_IN_B, J_WQ, J_WK, J_WV, J_WO, J_LN1S, J_LN1B,
  J_W1, J_B1, J_W2, J_B2, J_LN2S, J_LN2B,
  C_IN_S, C_IN_B, C_WQ, C_WK, C_WV, C_WO, C_LN1S, C_LN1B,
  C_W1, C_B1, C_W2, C_B2, C_LN2S, C_LN2B, C_PROJW, C_PROJB,
  C_CQ, C_CK, C_CV, C_CO, C_LNCS, C_LNCB,
  IN_PRJ, IN_START, N_INPUTS
};

extern "C" void kernel_launch(void* const* d_in, const int* in_sizes, int n_in,
                              void* d_out, int out_size, void* d_ws, size_t ws_size,
                              hipStream_t stream)
{
  (void)in_sizes; (void)n_in; (void)out_size; (void)ws_size;
  auto P = [&](int i) { return (const float*)d_in[i]; };

  float* out = (float*)d_out;
  float* outq_o = out;                 // [2,192,32,32]
  float* lik_o  = out + 393216;        // [2,192,32,32]
  float* cond_o = out + 786432;        // [2,384,32,32]
  float* zf_o   = out + 1572864;       // [2,192,32,32]

  // ---- workspace layout ----
  const size_t S = (size_t)16384 * D;  // rows*192
  float* X  = (float*)d_ws;            // residual stream
  float* Y  = X + S;                   // GEMM f32 outputs
  float* Qf = Y + S;
  float* Kf = Qf + S;
  float* Vf = Kf + S;
  float* Zj = Vf + S;                  // joint-encoder stream / decoder memory
  unsigned short* Xb  = (unsigned short*)(Zj + S);   // bf16 activation mirror
  unsigned short* Zjb = Xb + S;                      // bf16 of z_joint
  unsigned short* Ab  = Zjb + S;                     // bf16 attention output
  unsigned short* Hb  = Ab + S;                      // bf16 FFN hidden (FCH x DI)
  unsigned short* arena = Hb + (size_t)FCH * DI;     // bf16 transposed weights
  size_t cur = 0;

  auto ctw = [&](const float* src, int Kd, int Nd) -> const unsigned short* {
    unsigned short* dst = arena + cur;
    cur += (size_t)Kd * Nd;
    int n = Kd * Nd;
    cvt_transpose_kernel<<<(n + 255) / 256, 256, 0, stream>>>(src, dst, Kd, Nd);
    return dst;
  };

  const int WQK = D * D;
  const int W1S = D * DI, W2S = DI * D;

  // ---- convert all weights (transposed bf16) once per call ----
  const unsigned short *swq[6], *swk[6], *swv[6], *swo[6], *sw1[6], *sw2[6];
  for (int i = 0; i < 6; ++i) {
    swq[i] = ctw(P(SEP_WQ) + (size_t)i * WQK, D, D);
    swk[i] = ctw(P(SEP_WK) + (size_t)i * WQK, D, D);
    swv[i] = ctw(P(SEP_WV) + (size_t)i * WQK, D, D);
    swo[i] = ctw(P(SEP_WO) + (size_t)i * WQK, D, D);
    sw1[i] = ctw(P(SEP_W1) + (size_t)i * W1S, D, DI);
    sw2[i] = ctw(P(SEP_W2) + (size_t)i * W2S, DI, D);
  }
  const unsigned short* sprojt = ctw(P(SEP_PROJW), D, D);
  const unsigned short *jwq[4], *jwk[4], *jwv[4], *jwo[4], *jw1[4], *jw2[4];
  for (int i = 0; i < 4; ++i) {
    jwq[i] = ctw(P(J_WQ) + (size_t)i * WQK, D, D);
    jwk[i] = ctw(P(J_WK) + (size_t)i * WQK, D, D);
    jwv[i] = ctw(P(J_WV) + (size_t)i * WQK, D, D);
    jwo[i] = ctw(P(J_WO) + (size_t)i * WQK, D, D);
    jw1[i] = ctw(P(J_W1) + (size_t)i * W1S, D, DI);
    jw2[i] = ctw(P(J_W2) + (size_t)i * W2S, DI, D);
  }
  const unsigned short *cwq[5], *cwk[5], *cwv[5], *cwo[5];
  const unsigned short *ccq[5], *cck[5], *ccv[5], *cco[5], *cw1[5], *cw2[5];
  for (int i = 0; i < 5; ++i) {
    cwq[i] = ctw(P(C_WQ) + (size_t)i * WQK, D, D);
    cwk[i] = ctw(P(C_WK) + (size_t)i * WQK, D, D);
    cwv[i] = ctw(P(C_WV) + (size_t)i * WQK, D, D);
    cwo[i] = ctw(P(C_WO) + (size_t)i * WQK, D, D);
    ccq[i] = ctw(P(C_CQ) + (size_t)i * WQK, D, D);
    cck[i] = ctw(P(C_CK) + (size_t)i * WQK, D, D);
    ccv[i] = ctw(P(C_CV) + (size_t)i * WQK, D, D);
    cco[i] = ctw(P(C_CO) + (size_t)i * WQK, D, D);
    cw1[i] = ctw(P(C_W1) + (size_t)i * W1S, D, DI);
    cw2[i] = ctw(P(C_W2) + (size_t)i * W2S, DI, D);
  }
  const unsigned short* cprojt = ctw(P(C_PROJW), D, D);
  const unsigned short* prjt   = ctw(P(IN_PRJ), D, 384);

  // common layer body (self-attn + FFN). Xs = f32 stream, Xb = bf16 mirror.
  auto enc_layer = [&](float* Xs, int M, int B, int L, int causal,
                       const unsigned short* wq, const unsigned short* wk,
                       const unsigned short* wv, const unsigned short* wo,
                       const float* l1s, const float* l1b,
                       const unsigned short* w1, const float* b1,
                       const unsigned short* w2, const float* b2,
                       const float* l2s, const float* l2b) {
    gemm(stream, Xb, wq, nullptr, Qf, M, D, D, 0, 1.f);
    gemm(stream, Xb, wk, nullptr, Kf, M, D, D, 0, 1.f);
    gemm(stream, Xb, wv, nullptr, Vf, M, D, D, 0, 1.f);
    attn(stream, Qf, Kf, Vf, Ab, B, L, L, causal);
    gemm(stream, Ab, wo, nullptr, Y, M, D, D, 0, 1.f);
    ln(stream, Xs, Xb, Xs, Y, l1s, l1b, M, 0);
    for (int m0 = 0; m0 < M; m0 += FCH) {
      int r = (M - m0 < FCH) ? (M - m0) : FCH;
      gemm(stream, Xb + (size_t)m0 * D, w1, b1, Hb, r, DI, D, 1 | 2 | 4, 1.f);
      gemm(stream, Hb, w2, b2, Y + (size_t)m0 * D, r, D, DI, 1, 1.f);
    }
    ln(stream, Xs, Xb, Xs, Y, l2s, l2b, M, 0);
  };

  // ---- out_q = feature + noise - 0.5 ----
  ew_outq_kernel<<<(393216 + 255) / 256, 256, 0, stream>>>(P(IN_FEATURE), P(IN_NOISE),
                                                           outq_o, 393216);

  // ---- sep encoders over prev1 / prev2 ----
  const float* prevs[2] = { P(IN_PREV1), P(IN_PREV2) };
  for (int pv = 0; pv < 2; ++pv) {
    gather_overlap_kernel<<<(8192 * D + 255) / 256, 256, 0, stream>>>(prevs[pv], Xb);
    gemm(stream, Xb, sprojt, P(SEP_PROJB), Y, 8192, D, D, 1, 1.f);
    ln(stream, X, Xb, Y, nullptr, P(SEP_IN_S), P(SEP_IN_B), 8192, 64);
    for (int i = 0; i < 6; ++i) {
      enc_layer(X, 8192, 128, 64, 0, swq[i], swk[i], swv[i], swo[i],
                P(SEP_LN1S) + i * D, P(SEP_LN1B) + i * D,
                sw1[i], P(SEP_B1) + i * DI, sw2[i], P(SEP_B2) + i * D,
                P(SEP_LN2S) + i * D, P(SEP_LN2B) + i * D);
    }
    scatter_zsep_kernel<<<(8192 * D + 255) / 256, 256, 0, stream>>>(X, Zj, pv * 64);
  }

  // ---- joint encoder on Zj (M=16384, L=128) ----
  ln(stream, Zj, Xb, Zj, nullptr, P(J_IN_S), P(J_IN_B), 16384, 128);
  for (int i = 0; i < 4; ++i) {
    enc_layer(Zj, 16384, 128, 128, 0, jwq[i], jwk[i], jwv[i], jwo[i],
              P(J_LN1S) + i * D, P(J_LN1B) + i * D,
              jw1[i], P(J_B1) + i * DI, jw2[i], P(J_B2) + i * D,
              P(J_LN2S) + i * D, P(J_LN2B) + i * D);
  }
  // bf16 copy of z_joint for the decoder cross-attention K/V GEMMs
  cvt_flat_kernel<<<((int)S + 255) / 256, 256, 0, stream>>>(Zj, Zjb, (int)S);

  // ---- decoder (M=2048, L=16, memory = Zj) ----
  dec_tok_kernel<<<(2048 * D + 255) / 256, 256, 0, stream>>>(outq_o, P(IN_START), Xb);
  gemm(stream, Xb, cprojt, P(C_PROJB), Y, 2048, D, D, 1, 1.f);
  ln(stream, X, Xb, Y, nullptr, P(C_IN_S), P(C_IN_B), 2048, 16);

  for (int i = 0; i < 5; ++i) {
    // masked self-attention
    gemm(stream, Xb, cwq[i], nullptr, Qf, 2048, D, D, 0, 1.f);
    gemm(stream, Xb, cwk[i], nullptr, Kf, 2048, D, D, 0, 1.f);
    gemm(stream, Xb, cwv[i], nullptr, Vf, 2048, D, D, 0, 1.f);
    attn(stream, Qf, Kf, Vf, Ab, 128, 16, 16, 1);
    gemm(stream, Ab, cwo[i], nullptr, Y, 2048, D, D, 0, 1.f);
    ln(stream, X, Xb, X, Y, P(C_LN1S) + i * D, P(C_LN1B) + i * D, 2048, 0);
    // cross-attention to z_joint
    gemm(stream, Xb, ccq[i], nullptr, Qf, 2048, D, D, 0, 1.f);
    gemm(stream, Zjb, cck[i], nullptr, Kf, 16384, D, D, 0, 1.f);
    gemm(stream, Zjb, ccv[i], nullptr, Vf, 16384, D, D, 0, 1.f);
    attn(stream, Qf, Kf, Vf, Ab, 128, 16, 128, 0);
    gemm(stream, Ab, cco[i], nullptr, Y, 2048, D, D, 0, 1.f);
    ln(stream, X, Xb, X, Y, P(C_LNCS) + i * D, P(C_LNCB) + i * D, 2048, 0);
    // FFN
    gemm(stream, Xb, cw1[i], P(C_B1) + i * DI, Hb, 2048, DI, D, 1 | 2 | 4, 1.f);
    gemm(stream, Hb, cw2[i], P(C_B2) + i * D, Y, 2048, D, DI, 1, 1.f);
    ln(stream, X, Xb, X, Y, P(C_LN2S) + i * D, P(C_LN2B) + i * D, 2048, 0);
  }

  // ---- dec = z_cur @ prj * 192^-0.5 ; condition / z_cur_f / likelihood ----
  gemm(stream, Xb, prjt, nullptr, Y, 2048, 384, D, 0, 0.07216878364870323f);
  cond_scatter_kernel<<<(786432 + 255) / 256, 256, 0, stream>>>(Y, cond_o);
  zf_scatter_kernel<<<(393216 + 255) / 256, 256, 0, stream>>>(X, zf_o);
  lik_kernel<<<(393216 + 255) / 256, 256, 0, stream>>>(outq_o, cond_o, lik_o);
}
